// Performer_68255620268908
// MI455X (gfx1250) — matrix operand order repaired and checked by reading, not compiled, not run on hardware
//
#include <hip/hip_runtime.h>
#include <hip/hip_bf16.h>
#include <math.h>

// ---------------------------------------------------------------------------
// Performer/local-attention transformer, 4 layers, for gfx1250 (MI455X).
// All GEMMs run through v_wmma_f32_16x16x32_bf16 (bf16 x bf16 -> f32 acc).
// Wave tile is templated: up to 64x64 per wave (16 WMMA per 32-K step) for
// the large matmuls, 32-granular tiles for the 288-wide performer matrices.
// ---------------------------------------------------------------------------

typedef __bf16 bf16_t;
typedef __attribute__((ext_vector_type(16))) __bf16 bf16x16;
typedef __attribute__((ext_vector_type(8)))  __bf16 bf16x8;
typedef __attribute__((ext_vector_type(8)))  float  floatx8;

union FragA { bf16x16 v; bf16x8 h[2]; };

// ---- problem constants ----
#define BATCH   2
#define NSEQ    4096
#define DIM     768
#define NHEADS  12
#define DH      64
#define GH      6
#define LH      6
#define INNER   768
#define FFDIM   3072
#define NTOK    (BATCH*NSEQ)          // 8192
#define MR      266                   // performer features
#define MP      288                   // MR padded to 32
#define NBH     (BATCH*GH)            // 12 performer (b,head) batches; also local
#define CCH     16                    // chunks (4096/256)
#define QCH     256
#define KCH     512                   // 510 padded to 512
#define KROWS   4480                  // 127 + 4096 + 257 pad rows for K/V windows
#define DNQ     0.35355339059327373f  // 64^-0.25
#define RATIO   0.06131392670380837f  // 266^-0.5
#define NEGBIG  (-3.4028234663852886e+38f)

enum { F_BIAS = 1, F_ADD = 2, F_GELU = 4, F_WF32 = 8, F_WBF = 16 };

// ---------------------------------------------------------------------------
// Generic batched NT GEMM: C[M,N] (+)= A[M,K] * B[N,K]^T, bf16 in, f32 acc.
// One wave per (16*TM)x(16*TN) tile; TM*TN WMMAs per 32-wide K step.
// M % (16*TM) == 0, N % (16*TN) == 0, K % 32 == 0 guaranteed by padding.
// Batch offset for grid.z = z:  i1 = z/div, i2 = z%div (two-level batching).
// ---------------------------------------------------------------------------
template <int TM, int TN>
__global__ __launch_bounds__(32)
void gemm_bf16_nt(const bf16_t* __restrict__ A, const bf16_t* __restrict__ B,
                  float* __restrict__ C, bf16_t* __restrict__ Dbf,
                  const float* __restrict__ bias,
                  int K, long lda, long ldb, long ldc,
                  int div, long sA1, long sA2, long sB1, long sB2,
                  long sC1, long sC2, int flags)
{
    const int z  = blockIdx.z;
    const long i1 = z / div, i2 = z % div;
    A += i1 * sA1 + i2 * sA2;
    B += i1 * sB1 + i2 * sB2;
    const long coff = i1 * sC1 + i2 * sC2;

    const int lane = threadIdx.x;
    const int r    = lane & 15;
    const int hi   = lane >> 4;
    const long m0  = (long)blockIdx.y * (16 * TM);
    const long n0  = (long)blockIdx.x * (16 * TN);

    floatx8 acc[TM][TN];
#pragma unroll
    for (int i = 0; i < TM; ++i)
#pragma unroll
        for (int j = 0; j < TN; ++j) acc[i][j] = (floatx8){};

    // A frag (16x32 bf16): lane r (lo half) holds row r, K {kb..kb+7, kb+16..23};
    // hi half lanes hold K {kb+8..15, kb+24..31}.  Two 16B loads per lane.
    const bf16_t* pa[TM];
#pragma unroll
    for (int i = 0; i < TM; ++i) pa[i] = A + (m0 + 16 * i + r) * lda + hi * 8;
    // B frag (32x16): lane r holds B-column n0+16j+r i.e. row n0+16j+r of the
    // NT-stored B, K {kb..kb+7, kb+16..23} (+8 for hi lanes), same as A.
    const bf16_t* pb[TN];
#pragma unroll
    for (int j = 0; j < TN; ++j) pb[j] = B + (n0 + 16 * j + r) * ldb + hi * 8;

    for (int kb = 0; kb < K; kb += 32) {
        FragA a[TM];
        bf16x16 b[TN];
#pragma unroll
        for (int i = 0; i < TM; ++i) {
            a[i].h[0] = *(const bf16x8*)(pa[i] + kb);
            a[i].h[1] = *(const bf16x8*)(pa[i] + kb + 16);
        }
#pragma unroll
        for (int j = 0; j < TN; ++j) b[j] = __builtin_shufflevector(*(const bf16x8*)(pb[j] + kb), *(const bf16x8*)(pb[j] + kb + 16), 0, 1, 2, 3, 4, 5, 6, 7, 8, 9, 10, 11, 12, 13, 14, 15);
        if (kb + 64 < K) {              // speculative prefetch, 2 steps ahead
#pragma unroll
            for (int i = 0; i < TM; ++i) __builtin_prefetch(pa[i] + kb + 64, 0, 0);
#pragma unroll
            for (int j = 0; j < TN; ++j) __builtin_prefetch(pb[j] + kb + 64, 0, 0);
        }
#pragma unroll
        for (int i = 0; i < TM; ++i)
#pragma unroll
            for (int j = 0; j < TN; ++j)
                acc[i][j] = __builtin_amdgcn_wmma_f32_16x16x32_bf16(
                    false, a[i].v, false, b[j], (short)0, acc[i][j], false, false);
    }

    // C/D layout: VGPR e, lanes 0-15 -> M = e, lanes 16-31 -> M = e + 8.
#pragma unroll
    for (int it = 0; it < TM; ++it) {
#pragma unroll
        for (int jt = 0; jt < TN; ++jt) {
            const long col = n0 + jt * 16 + r;
            const float bval = (flags & F_BIAS) ? bias[col] : 0.0f;
#pragma unroll
            for (int e = 0; e < 8; ++e) {
                const long row = m0 + it * 16 + hi * 8 + e;
                float v = acc[it][jt][e] + bval;
                const long ci = coff + row * ldc + col;
                if (flags & F_ADD)  v += C[ci];
                if (flags & F_GELU) v = 0.5f * v * (1.0f + erff(v * 0.70710678118654752f));
                if (flags & F_WF32) C[ci]   = v;
                if (flags & F_WBF)  Dbf[ci] = (bf16_t)v;
            }
        }
    }
}

// ---------------------------------------------------------------------------
template <int OP>  // 0 = sum, 1 = max
__device__ inline float block_reduce(float v, float* sh) {
    const int t = threadIdx.x;
    sh[t] = v;
    __syncthreads();
    for (int s = blockDim.x >> 1; s > 0; s >>= 1) {
        if (t < s) sh[t] = OP ? fmaxf(sh[t], sh[t + s]) : sh[t] + sh[t + s];
        __syncthreads();
    }
    float r = sh[0];
    __syncthreads();
    return r;
}

// ---- LayerNorm (fp32 in -> bf16 out) --------------------------------------
__global__ __launch_bounds__(256)
void ln_kernel(const float* __restrict__ x, const float* __restrict__ g,
               const float* __restrict__ b, bf16_t* __restrict__ out)
{
    __shared__ float sh[256];
    const long row = blockIdx.x;
    const float* xr = x + row * DIM;
    float s = 0.0f;
    for (int i = threadIdx.x; i < DIM; i += 256) s += xr[i];
    const float mean = block_reduce<0>(s, sh) * (1.0f / DIM);
    float v = 0.0f;
    for (int i = threadIdx.x; i < DIM; i += 256) {
        float d = xr[i] - mean;
        v += d * d;
    }
    const float rstd = rsqrtf(block_reduce<0>(v, sh) * (1.0f / DIM) + 1e-5f);
    for (int i = threadIdx.x; i < DIM; i += 256)
        out[row * DIM + i] = (bf16_t)((xr[i] - mean) * rstd * g[i] + b[i]);
}

// ---- weight transpose+convert: src f32 [K,N] -> dst bf16 [N,K] ------------
__global__ __launch_bounds__(256)
void transpose_f32_bf16(const float* __restrict__ src, bf16_t* __restrict__ dst,
                        int K, int N)
{
    const long idx = (long)blockIdx.x * 256 + threadIdx.x;
    if (idx >= (long)K * N) return;
    const long k = idx / N, n = idx % N;
    dst[n * (long)K + k] = (bf16_t)src[idx];
}

// ---- proj -> bf16 [MP,DH], rows >= MR zeroed -------------------------------
__global__ __launch_bounds__(256)
void proj_conv_kernel(const float* __restrict__ proj, bf16_t* __restrict__ dst)
{
    const int idx = blockIdx.x * 256 + threadIdx.x;
    if (idx >= MP * DH) return;
    const int m = idx / DH;
    dst[idx] = (m < MR) ? (bf16_t)proj[idx] : (bf16_t)0.0f;
}

// ---- performer prep: scale q/k, diag, transpose v --------------------------
__global__ __launch_bounds__(64)
void prep_global_kernel(const float* __restrict__ qkv, bf16_t* __restrict__ qsb,
                        bf16_t* __restrict__ ksb, bf16_t* __restrict__ vtb,
                        float* __restrict__ diagq, float* __restrict__ diagk)
{
    __shared__ float rq[64], rk[64];
    const int bhn = blockIdx.x;          // bh*4096 + n
    const int bh = bhn >> 12, n = bhn & 4095;
    const int b = bh / GH, h = bh % GH;  // global heads are heads [0,6)
    const int d = threadIdx.x;
    const float* base = qkv + ((long)(b * NSEQ + n)) * (3 * INNER) + h * DH + d;
    const float q = base[0], k = base[INNER], v = base[2 * INNER];
    qsb[(long)bhn * DH + d] = (bf16_t)(q * DNQ);
    ksb[(long)bhn * DH + d] = (bf16_t)(k * DNQ);
    vtb[(long)bh * DH * NSEQ + (long)d * NSEQ + n] = (bf16_t)v;
    rq[d] = q * q; rk[d] = k * k;
    __syncthreads();
    for (int s = 32; s > 0; s >>= 1) {
        if (d < s) { rq[d] += rq[d + s]; rk[d] += rk[d + s]; }
        __syncthreads();
    }
    if (d == 0) {
        diagq[bhn] = 0.5f * DNQ * DNQ * rq[0];
        diagk[bhn] = 0.5f * DNQ * DNQ * rk[0];
    }
}

// ---- local prep: scale q, RoPE q/k, scatter k/v into padded buffers --------
__global__ __launch_bounds__(64)
void prep_local_kernel(const float* __restrict__ qkv, bf16_t* __restrict__ qlb,
                       bf16_t* __restrict__ kpad, bf16_t* __restrict__ vpadT)
{
    __shared__ float sq[64], sk[64];
    const int bhn = blockIdx.x;
    const int bh = bhn >> 12, n = bhn & 4095;
    const int b = bh / LH, h = GH + (bh % LH);   // local heads are heads [6,12)
    const int d = threadIdx.x;
    const float* base = qkv + ((long)(b * NSEQ + n)) * (3 * INNER) + h * DH + d;
    const float qs = base[0] * 0.125f;           // 64^-0.5
    const float k  = base[INNER];
    const float v  = base[2 * INNER];
    sq[d] = qs; sk[d] = k;
    __syncthreads();
    const float inv = powf(10000.0f, -(float)(2 * (d & 31)) / (float)DH);
    const float ang = (float)n * inv;
    const float c = cosf(ang), s = sinf(ang);
    const float qo = (d < 32) ? (qs * c - sq[d + 32] * s) : (qs * c + sq[d - 32] * s);
    const float ko = (d < 32) ? (k  * c - sk[d + 32] * s) : (k  * c + sk[d - 32] * s);
    qlb[(long)bhn * DH + d] = (bf16_t)qo;
    kpad[(long)bh * KROWS * DH + (long)(127 + n) * DH + d] = (bf16_t)ko;
    vpadT[(long)bh * DH * KROWS + (long)d * KROWS + 127 + n] = (bf16_t)v;
}

// ---- per-head global max over dd (keys) ------------------------------------
__global__ __launch_bounds__(256)
void kmax_kernel(const float* __restrict__ dd, float* __restrict__ headmax)
{
    __shared__ float sh[256];
    const int bh = blockIdx.x;
    const float* base = dd + (long)bh * NSEQ * MP;
    float m = NEGBIG;
    for (long i = threadIdx.x; i < (long)NSEQ * MR; i += 256) {
        const long n = i / MR, mm = i % MR;
        m = fmaxf(m, base[n * MP + mm]);
    }
    m = block_reduce<1>(m, sh);
    if (threadIdx.x == 0) headmax[bh] = m;
}

// ---- key feature map, writes kp f32 in place + transposed bf16 -------------
__global__ __launch_bounds__(256)
void kexp_kernel(float* __restrict__ dd, const float* __restrict__ diagk,
                 const float* __restrict__ headmax, bf16_t* __restrict__ kptb)
{
    const int bhn = blockIdx.x;
    const int bh = bhn >> 12, n = bhn & 4095;
    const long base = (long)bhn * MP;
    const float dg = diagk[bhn], mx = headmax[bh];
    for (int m = threadIdx.x; m < MP; m += 256) {
        float val = 0.0f;
        if (m < MR) val = RATIO * (expf(dd[base + m] - dg - mx) + 1e-4f);
        dd[base + m] = val;
        kptb[(long)bh * MP * NSEQ + (long)m * NSEQ + n] = (bf16_t)val;
    }
}

// ---- ksum[m] = sum_n kp[n,m] ------------------------------------------------
__global__ __launch_bounds__(256)
void ksum_kernel(const float* __restrict__ kp, float* __restrict__ ksum)
{
    __shared__ float sh[256];
    const int bh = blockIdx.x / MP, m = blockIdx.x % MP;
    float s = 0.0f;
    for (int n = threadIdx.x; n < NSEQ; n += 256)
        s += kp[(long)bh * NSEQ * MP + (long)n * MP + m];
    s = block_reduce<0>(s, sh);
    if (threadIdx.x == 0) ksum[blockIdx.x] = s;
}

// ---- query feature map (per-row max), writes qp f32 in place + bf16 --------
__global__ __launch_bounds__(256)
void qexp_kernel(float* __restrict__ dd, const float* __restrict__ diagq,
                 bf16_t* __restrict__ qpb)
{
    __shared__ float sh[256];
    const int bhn = blockIdx.x;
    const long base = (long)bhn * MP;
    float m = NEGBIG;
    for (int mm = threadIdx.x; mm < MR; mm += 256) m = fmaxf(m, dd[base + mm]);
    const float mx = block_reduce<1>(m, sh);
    const float dg = diagq[bhn];
    for (int mm = threadIdx.x; mm < MP; mm += 256) {
        float val = 0.0f;
        if (mm < MR) val = RATIO * (expf(dd[base + mm] - dg - mx) + 1e-4f);
        dd[base + mm] = val;
        qpb[base + mm] = (bf16_t)val;
    }
}

// ---- d_inv[n] = 1 / (qp[n,:] . ksum) ---------------------------------------
__global__ __launch_bounds__(256)
void dinv_kernel(const float* __restrict__ qp, const float* __restrict__ ksum,
                 float* __restrict__ dinv)
{
    __shared__ float sh[256];
    const int bhn = blockIdx.x;
    const int bh = bhn >> 12;
    float s = 0.0f;
    for (int m = threadIdx.x; m < MP; m += 256)   // pads are exact zeros
        s += qp[(long)bhn * MP + m] * ksum[bh * MP + m];
    s = block_reduce<0>(s, sh);
    if (threadIdx.x == 0) dinv[bhn] = 1.0f / s;
}

// ---- ctx [MP,DH] f32 -> ctxT bf16 [DH,MP] ----------------------------------
__global__ __launch_bounds__(256)
void ctxT_kernel(const float* __restrict__ ctx, bf16_t* __restrict__ ctxTb)
{
    const long idx = (long)blockIdx.x * 256 + threadIdx.x;
    if (idx >= (long)NBH * MP * DH) return;
    const long bh = idx / (MP * DH);
    const long rem = idx % (MP * DH);
    const long m = rem / DH, d = rem % DH;
    ctxTb[bh * DH * MP + d * MP + m] = (bf16_t)ctx[idx];
}

// ---- masked row softmax for local attention --------------------------------
__global__ __launch_bounds__(256)
void softmax_local_kernel(const float* __restrict__ S, bf16_t* __restrict__ P)
{
    __shared__ float sh[256];
    const int rowid = blockIdx.x;
    const int z = rowid >> 8;        // (bh, chunk) batch
    const int i = rowid & 255;       // query row in chunk
    const int c = z & 15;            // chunk index
    const float* Sr = S + (long)rowid * KCH;
    bf16_t* Pr = P + (long)rowid * KCH;

    float v[2];
    float mloc = NEGBIG;
#pragma unroll
    for (int t = 0; t < 2; ++t) {
        const int j = threadIdx.x + t * 256;
        const int kg = c * QCH + j - 127;
        const bool valid = (j >= i) && (j <= i + 254) && (j < 510) &&
                           (kg >= 0) && (kg < NSEQ);
        v[t] = valid ? Sr[j] : NEGBIG;
        mloc = fmaxf(mloc, v[t]);
    }
    const float mx = block_reduce<1>(mloc, sh);
    float p[2];
    float sloc = 0.0f;
#pragma unroll
    for (int t = 0; t < 2; ++t) {
        p[t] = expf(v[t] - mx);      // masked lanes underflow to exact 0
        sloc += p[t];
    }
    const float inv = 1.0f / block_reduce<0>(sloc, sh);
#pragma unroll
    for (int t = 0; t < 2; ++t)
        Pr[threadIdx.x + t * 256] = (bf16_t)(p[t] * inv);
}

// ---- assemble attention output (bf16) for the output projection ------------
__global__ __launch_bounds__(768)
void pack_o_kernel(const float* __restrict__ og, const float* __restrict__ dinv,
                   const float* __restrict__ zb, bf16_t* __restrict__ obf)
{
    const int t = blockIdx.x;            // token
    const int b = t >> 12, n = t & 4095;
    const int idx = threadIdx.x;
    const int h = idx >> 6, d = idx & 63;
    float val;
    if (h < GH) {
        const int bh = b * GH + h;
        val = og[((long)bh * NSEQ + n) * DH + d] * dinv[(long)bh * NSEQ + n];
    } else {
        const int bh = b * LH + (h - GH);
        val = zb[((long)bh * NSEQ + n) * DH + d];
    }
    obf[(long)t * INNER + idx] = (bf16_t)val;
}

// ---------------------------------------------------------------------------
static inline void gemm(hipStream_t s, int TM, int TN,
                        const bf16_t* A, const bf16_t* B,
                        float* C, bf16_t* Dbf, const float* bias,
                        int M, int N, int K, long lda, long ldb, long ldc,
                        int batches, int div, long sA1, long sA2,
                        long sB1, long sB2, long sC1, long sC2, int flags)
{
    dim3 g(N / (16 * TN), M / (16 * TM), batches), blk(32);
    if (TM == 4 && TN == 4)
        gemm_bf16_nt<4, 4><<<g, blk, 0, s>>>(A, B, C, Dbf, bias, K, lda, ldb, ldc,
                                             div, sA1, sA2, sB1, sB2, sC1, sC2, flags);
    else if (TM == 4 && TN == 2)
        gemm_bf16_nt<4, 2><<<g, blk, 0, s>>>(A, B, C, Dbf, bias, K, lda, ldb, ldc,
                                             div, sA1, sA2, sB1, sB2, sC1, sC2, flags);
    else if (TM == 2 && TN == 4)
        gemm_bf16_nt<2, 4><<<g, blk, 0, s>>>(A, B, C, Dbf, bias, K, lda, ldb, ldc,
                                             div, sA1, sA2, sB1, sB2, sC1, sC2, flags);
    else
        gemm_bf16_nt<2, 2><<<g, blk, 0, s>>>(A, B, C, Dbf, bias, K, lda, ldb, ldc,
                                             div, sA1, sA2, sB1, sB2, sC1, sC2, flags);
}

extern "C" void kernel_launch(void* const* d_in, const int* in_sizes, int n_in,
                              void* d_out, int out_size, void* d_ws, size_t ws_size,
                              hipStream_t stream)
{
    const float* x_in = (const float*)d_in[0];
    const float* ln1g = (const float*)d_in[1];
    const float* ln1b = (const float*)d_in[2];
    const float* wq   = (const float*)d_in[3];
    const float* bq   = (const float*)d_in[4];
    const float* wk   = (const float*)d_in[5];
    const float* bk   = (const float*)d_in[6];
    const float* wv   = (const float*)d_in[7];
    const float* bv   = (const float*)d_in[8];
    const float* wo   = (const float*)d_in[9];
    const float* bo   = (const float*)d_in[10];
    const float* ln2g = (const float*)d_in[11];
    const float* ln2b = (const float*)d_in[12];
    const float* w1   = (const float*)d_in[13];
    const float* b1   = (const float*)d_in[14];
    const float* w2   = (const float*)d_in[15];
    const float* b2   = (const float*)d_in[16];
    const float* proj = (const float*)d_in[17];
    float* x = (float*)d_out;

    // ---- workspace bump allocator (deterministic layout each call) ----
    char* ws = (char*)d_ws;
    size_t off = 0;
    auto alloc = [&](size_t bytes) -> char* {
        char* p = ws + off;
        off = (off + bytes + 255) & ~(size_t)255;
        return p;
    };
    bf16_t* hb     = (bf16_t*)alloc((size_t)NTOK * DIM * 2);
    float*  qkv    = (float*) alloc((size_t)NTOK * 3 * INNER * 4);
    bf16_t* wqkvT  = (bf16_t*)alloc((size_t)3 * INNER * DIM * 2);
    bf16_t* woT    = (bf16_t*)alloc((size_t)DIM * INNER * 2);
    bf16_t* w1T    = (bf16_t*)alloc((size_t)FFDIM * DIM * 2);
    bf16_t* w2T    = (bf16_t*)alloc((size_t)DIM * FFDIM * 2);
    float*  bqkv   = (float*) alloc((size_t)3 * INNER * 4);
    bf16_t* projb  = (bf16_t*)alloc((size_t)MP * DH * 2);
    bf16_t* qsb    = (bf16_t*)alloc((size_t)NBH * NSEQ * DH * 2);
    bf16_t* ksb    = (bf16_t*)alloc((size_t)NBH * NSEQ * DH * 2);
    float*  diagq  = (float*) alloc((size_t)NBH * NSEQ * 4);
    float*  diagk  = (float*) alloc((size_t)NBH * NSEQ * 4);
    bf16_t* vtb    = (bf16_t*)alloc((size_t)NBH * DH * NSEQ * 2);
    float*  dd     = (float*) alloc((size_t)NBH * NSEQ * MP * 4);
    bf16_t* qpb    = (bf16_t*)alloc((size_t)NBH * NSEQ * MP * 2);
    bf16_t* kptb   = (bf16_t*)alloc((size_t)NBH * MP * NSEQ * 2);
    float*  hmax   = (float*) alloc((size_t)NBH * 4);
    float*  ksum   = (float*) alloc((size_t)NBH * MP * 4);
    float*  dinv   = (float*) alloc((size_t)NBH * NSEQ * 4);
    float*  ctx    = (float*) alloc((size_t)NBH * MP * DH * 4);
    bf16_t* ctxTb  = (bf16_t*)alloc((size_t)NBH * DH * MP * 2);
    float*  og     = (float*) alloc((size_t)NBH * NSEQ * DH * 4);
    bf16_t* qlb    = (bf16_t*)alloc((size_t)NBH * NSEQ * DH * 2);
    bf16_t* kpad   = (bf16_t*)alloc((size_t)NBH * KROWS * DH * 2);
    bf16_t* vpadT  = (bf16_t*)alloc((size_t)NBH * DH * KROWS * 2);
    float*  Sbuf   = (float*) alloc((size_t)NBH * CCH * QCH * KCH * 4);
    bf16_t* Pbuf   = (bf16_t*)alloc((size_t)NBH * CCH * QCH * KCH * 2);
    float*  zb     = (float*) alloc((size_t)NBH * NSEQ * DH * 4);
    bf16_t* obf    = (bf16_t*)alloc((size_t)NTOK * INNER * 2);
    bf16_t* ffb    = (bf16_t*)alloc((size_t)NTOK * FFDIM * 2);

    // running x lives in d_out
    hipMemcpyAsync(x, x_in, (size_t)NTOK * DIM * 4, hipMemcpyDeviceToDevice, stream);
    // zero padded K/V window buffers once (data rows rewritten each layer)
    hipMemsetAsync(kpad,  0, (size_t)NBH * KROWS * DH * 2, stream);
    hipMemsetAsync(vpadT, 0, (size_t)NBH * DH * KROWS * 2, stream);

    for (int L = 0; L < 4; ++L) {
        const long wqo = (long)L * DIM * INNER;
        // ---- weights -> bf16 NT layout ----
        {
            const int nblk = (DIM * INNER + 255) / 256;
            transpose_f32_bf16<<<nblk, 256, 0, stream>>>(wq + wqo, wqkvT, DIM, INNER);
            transpose_f32_bf16<<<nblk, 256, 0, stream>>>(wk + wqo, wqkvT + (long)INNER * DIM, DIM, INNER);
            transpose_f32_bf16<<<nblk, 256, 0, stream>>>(wv + wqo, wqkvT + 2L * INNER * DIM, DIM, INNER);
            transpose_f32_bf16<<<nblk, 256, 0, stream>>>(wo + wqo, woT, INNER, DIM);
            const int nblk1 = (DIM * FFDIM + 255) / 256;
            transpose_f32_bf16<<<nblk1, 256, 0, stream>>>(w1 + (long)L * DIM * FFDIM, w1T, DIM, FFDIM);
            transpose_f32_bf16<<<nblk1, 256, 0, stream>>>(w2 + (long)L * FFDIM * DIM, w2T, FFDIM, DIM);
            proj_conv_kernel<<<(MP * DH + 255) / 256, 256, 0, stream>>>(proj + (long)L * MR * DH, projb);
            hipMemcpyAsync(bqkv,             bq + L * INNER, INNER * 4, hipMemcpyDeviceToDevice, stream);
            hipMemcpyAsync(bqkv + INNER,     bk + L * INNER, INNER * 4, hipMemcpyDeviceToDevice, stream);
            hipMemcpyAsync(bqkv + 2 * INNER, bv + L * INNER, INNER * 4, hipMemcpyDeviceToDevice, stream);
        }

        // ---- LN1 -> hb ----
        ln_kernel<<<NTOK, 256, 0, stream>>>(x, ln1g + L * DIM, ln1b + L * DIM, hb);

        // ---- fused QKV projection: [8192,768] x [2304,768]^T ----
        gemm(stream, 4, 4, hb, wqkvT, qkv, nullptr, bqkv, NTOK, 3 * INNER, DIM,
             DIM, DIM, 3 * INNER, 1, 1, 0, 0, 0, 0, 0, 0, F_BIAS | F_WF32);

        // ---- attention prep ----
        prep_global_kernel<<<NBH * NSEQ, 64, 0, stream>>>(qkv, qsb, ksb, vtb, diagq, diagk);
        prep_local_kernel<<<NBH * NSEQ, 64, 0, stream>>>(qkv, qlb, kpad, vpadT);

        // ---- performer, key path first (dd buffer reused for queries) ----
        gemm(stream, 4, 2, ksb, projb, dd, nullptr, nullptr, NSEQ, MP, DH,
             DH, DH, MP, NBH, 1, (long)NSEQ * DH, 0, 0, 0, (long)NSEQ * MP, 0, F_WF32);
        kmax_kernel<<<NBH, 256, 0, stream>>>(dd, hmax);
        kexp_kernel<<<NBH * NSEQ, 256, 0, stream>>>(dd, diagk, hmax, kptb);
        ksum_kernel<<<NBH * MP, 256, 0, stream>>>(dd, ksum);
        // ctx = kp^T v : [288,4096] x [64,4096]^T
        gemm(stream, 2, 4, kptb, vtb, ctx, nullptr, nullptr, MP, DH, NSEQ,
             NSEQ, NSEQ, DH, NBH, 1, (long)MP * NSEQ, 0, (long)DH * NSEQ, 0,
             (long)MP * DH, 0, F_WF32);
        ctxT_kernel<<<(int)(((long)NBH * MP * DH + 255) / 256), 256, 0, stream>>>(ctx, ctxTb);
        // query path
        gemm(stream, 4, 2, qsb, projb, dd, nullptr, nullptr, NSEQ, MP, DH,
             DH, DH, MP, NBH, 1, (long)NSEQ * DH, 0, 0, 0, (long)NSEQ * MP, 0, F_WF32);
        qexp_kernel<<<NBH * NSEQ, 256, 0, stream>>>(dd, diagq, qpb);
        dinv_kernel<<<NBH * NSEQ, 256, 0, stream>>>(dd, ksum, dinv);
        // out_g = qp @ ctx : [4096,288] x [64,288]^T (sB1 = 0: ctxT not batched; DH*MP goes to unused sA2)
        gemm(stream, 4, 4, qpb, ctxTb, og, nullptr, nullptr, NSEQ, DH, MP,
             MP, MP, DH, NBH, 1, (long)NSEQ * MP, (long)DH * MP, 0, 0,
             (long)NSEQ * DH, 0, F_WF32);

        // ---- local window attention ----
        // scores: per (bh, chunk): [256,64] x [512,64]^T
        gemm(stream, 4, 4, qlb, kpad, Sbuf, nullptr, nullptr, QCH, KCH, DH,
             DH, DH, KCH, NBH * CCH, CCH,
             (long)NSEQ * DH, (long)QCH * DH,
             (long)KROWS * DH, (long)QCH * DH,
             (long)CCH * QCH * KCH, (long)QCH * KCH, F_WF32);
        softmax_local_kernel<<<NBH * CCH * QCH, 256, 0, stream>>>(Sbuf, Pbuf);
        // z = P @ V : [256,512] x [64,512(strided)]^T
        gemm(stream, 4, 4, Pbuf, vpadT, zb, nullptr, nullptr, QCH, DH, KCH,
             KCH, KROWS, DH, NBH * CCH, CCH,
             (long)CCH * QCH * KCH, (long)QCH * KCH,
             (long)DH * KROWS, (long)QCH,
             (long)NSEQ * DH, (long)QCH * DH, F_WF32);

        // ---- combine heads, output projection (residual add) ----
        pack_o_kernel<<<NTOK, 768, 0, stream>>>(og, dinv, zb, obf);
        gemm(stream, 4, 4, obf, woT, x, nullptr, bo + L * DIM, NTOK, DIM, INNER,
             INNER, INNER, DIM, 1, 1, 0, 0, 0, 0, 0, 0, F_BIAS | F_ADD | F_WF32);

        // ---- FF block ----
        ln_kernel<<<NTOK, 256, 0, stream>>>(x, ln2g + L * DIM, ln2b + L * DIM, hb);
        gemm(stream, 4, 4, hb, w1T, nullptr, ffb, b1 + L * FFDIM, NTOK, FFDIM, DIM,
             DIM, DIM, FFDIM, 1, 1, 0, 0, 0, 0, 0, 0, F_BIAS | F_GELU | F_WBF);
        gemm(stream, 4, 4, ffb, w2T, x, nullptr, b2 + L * DIM, NTOK, DIM, FFDIM,
             FFDIM, FFDIM, DIM, 1, 1, 0, 0, 0, 0, 0, 0, F_BIAS | F_ADD | F_WF32);
    }
    (void)in_sizes; (void)n_in; (void)out_size; (void)ws_size;
}
